// VectorQuantizer_20083267076905
// MI455X (gfx1250) — matrix operand order repaired and checked by reading, not compiled, not run on hardware
//
#include <hip/hip_runtime.h>
#include <stdint.h>

// ---------------------------------------------------------------------------
// VQ-VAE vector quantizer for gfx1250 (MI455X)
//   wave32, v_wmma_f32_16x16x32_bf16, async global->LDS codebook staging
//   N = 262144 rows, D = 64, K = 1024 codes
// distances argmin uses  d(k) = |c_k|^2 - 2 x.c_k   (|x|^2 dropped: row-const)
// losses computed exactly in f32 from gathered codebook rows; loss reduction
// is atomics-free (deterministic per-block partials) for graph-replay safety.
// ---------------------------------------------------------------------------

typedef __attribute__((ext_vector_type(16))) __bf16 v16bf;
typedef __attribute__((ext_vector_type(8)))  float  v8f;

#define NROWS  262144   // 64*64*64
#define DDIM   64
#define KCODES 1024
#define NBLOCK (NROWS / 256)   // 1024 main-kernel blocks

// ws layout (bytes) -- first 135168 bytes are mirrored into LDS by vq_main:
//   [0,      131072)  bf16 codebook, pre-swizzled to WMMA B tile layout
//   [131072, 135168)  cnorm[1024]  f32
//   [135168, 139264)  counts[1024] f32
//   [139264, 143360)  lossPart[1024] f32 (one per main block, no atomics)
#define WS_CNORM   131072
#define WS_COUNTS  135168
#define WS_LOSSP   139264
#define LDS_BYTES  135168   // codebook tiles + cnorm

__device__ __forceinline__ unsigned short f2bf(float f) {
    unsigned int u = __float_as_uint(f);
    unsigned int r = u + 0x7FFFu + ((u >> 16) & 1u);   // round-to-nearest-even
    return (unsigned short)(r >> 16);
}

// ---- prep 1: codebook f32 -> bf16 in WMMA B-matrix lane layout -------------
__global__ void vq_prep_convert(const float* __restrict__ cb,
                                unsigned short* __restrict__ cbb) {
    int i = blockIdx.x * blockDim.x + threadIdx.x;     // 65536 elements
    int k = i >> 6, d = i & 63;
    int t = k >> 4, n = k & 15;
    int c = d >> 5, dl = d & 31;
    int lane = n + ((dl & 16) ? 16 : 0);
    int e = dl & 15;
    int off = (((t * 2 + c) * 32) + lane) * 16 + e;
    cbb[off] = f2bf(cb[i]);
}

// ---- prep 2: |c_k|^2, zero counts (deterministic per call) -----------------
__global__ void vq_prep_norm(const float* __restrict__ cb,
                             float* __restrict__ cnorm,
                             float* __restrict__ counts) {
    int k = blockIdx.x * blockDim.x + threadIdx.x;     // 1024 codes
    const float* r = cb + (size_t)k * DDIM;
    float s = 0.f;
#pragma unroll
    for (int d = 0; d < DDIM; ++d) { float v = r[d]; s += v * v; }
    cnorm[k]  = s;
    counts[k] = 0.f;
}

// ---- main: 8 waves x 32 rows = 256 rows/block; codebook staged in LDS ------
__global__ __launch_bounds__(256) void vq_main(
        const float* __restrict__ x, const float* __restrict__ cb,
        const char* __restrict__ wsimg,      // ws[0 .. LDS_BYTES): cbb + cnorm
        float* __restrict__ outq, float* __restrict__ outtok,
        float* __restrict__ counts, float* __restrict__ lossPart) {
    __shared__ __align__(16) unsigned char smem[LDS_BYTES];
    __shared__ float wsum[8];

    const int tid  = threadIdx.x;
    const int lane = tid & 31;
    const int wid  = tid >> 5;
    const int half = lane >> 4;        // 0: acc slots = rows 0-7, 1: rows 8-15
    const int m    = lane & 15;        // A row within tile / B column
    const int rowBase = blockIdx.x * 256 + wid * 32;

    // --- async-copy codebook image (swizzled bf16 tiles + cnorm) into LDS ---
    // 135168 B = 33 * 256 threads * 16 B; tracked with ASYNCcnt.
    {
        unsigned lbase = (unsigned)(uintptr_t)&smem[0];
#pragma unroll
        for (int i = 0; i < LDS_BYTES / (256 * 16); ++i) {
            unsigned off = ((unsigned)tid + (unsigned)i * 256u) * 16u;
            unsigned laddr = lbase + off;
            const void* g = wsimg + off;
            asm volatile("global_load_async_to_lds_b128 %0, %1, off"
                         :: "v"(laddr), "v"(g) : "memory");
        }
        asm volatile("s_wait_asynccnt 0" ::: "memory");
        __syncthreads();
    }
    const v16bf* lcb = (const v16bf*)smem;
    const float* lcn = (const float*)(smem + WS_CNORM);

    // --- build two A fragments (rows 0-15 = Lo, 16-31 = Hi), bf16 ----------
    // A layout (matches B packing): lanes 0-15 hold K in {0..15}(+32),
    //                               lanes 16-31 hold K in {16..31}(+32)
    union U { v16bf v; unsigned short u[16]; };
    U aLo[2], aHi[2];
    const int s0 = half ? 16 : 0;
    const int s1 = half ? 24 : 8;
#pragma unroll
    for (int g = 0; g < 2; ++g) {                     // 0 = Lo, 1 = Hi
        U* ua = g ? aHi : aLo;
        const float* xr = x + (size_t)(rowBase + g * 16 + m) * DDIM;
#pragma unroll
        for (int c = 0; c < 2; ++c) {
            const float4* p0 = (const float4*)(xr + c * 32 + s0);
            const float4* p1 = (const float4*)(xr + c * 32 + s1);
            float4 f0 = p0[0], f1 = p0[1], g0 = p1[0], g1 = p1[1];
            ua[c].u[0]  = f2bf(f0.x); ua[c].u[1]  = f2bf(f0.y);
            ua[c].u[2]  = f2bf(f0.z); ua[c].u[3]  = f2bf(f0.w);
            ua[c].u[4]  = f2bf(f1.x); ua[c].u[5]  = f2bf(f1.y);
            ua[c].u[6]  = f2bf(f1.z); ua[c].u[7]  = f2bf(f1.w);
            ua[c].u[8]  = f2bf(g0.x); ua[c].u[9]  = f2bf(g0.y);
            ua[c].u[10] = f2bf(g0.z); ua[c].u[11] = f2bf(g0.w);
            ua[c].u[12] = f2bf(g1.x); ua[c].u[13] = f2bf(g1.y);
            ua[c].u[14] = f2bf(g1.z); ua[c].u[15] = f2bf(g1.w);
        }
    }

    float bestLo[8], bestHi[8];
    int   bidxLo[8], bidxHi[8];
#pragma unroll
    for (int j = 0; j < 8; ++j) {
        bestLo[j] = 3.4e38f; bidxLo[j] = 0x7fffffff;
        bestHi[j] = 3.4e38f; bidxHi[j] = 0x7fffffff;
    }

    // --- 64 tiles of 16 codes; B from LDS; 4 WMMAs per tile ----------------
    // Per-lane candidate index n is strictly increasing over t, so a strict
    // '<' update already keeps the FIRST minimum (reference argmin); the
    // tie-break compare is only needed in the cross-lane reduction below.
#pragma unroll 2
    for (int t = 0; t < 64; ++t) {
        v16bf b0 = lcb[(t * 2 + 0) * 32 + lane];
        v16bf b1 = lcb[(t * 2 + 1) * 32 + lane];
        v8f accLo = {}, accHi = {};
        accLo = __builtin_amdgcn_wmma_f32_16x16x32_bf16(
                    false, aLo[0].v, false, b0, (short)0, accLo, false, false);
        accLo = __builtin_amdgcn_wmma_f32_16x16x32_bf16(
                    false, aLo[1].v, false, b1, (short)0, accLo, false, false);
        accHi = __builtin_amdgcn_wmma_f32_16x16x32_bf16(
                    false, aHi[0].v, false, b0, (short)0, accHi, false, false);
        accHi = __builtin_amdgcn_wmma_f32_16x16x32_bf16(
                    false, aHi[1].v, false, b1, (short)0, accHi, false, false);
        const int n = t * 16 + m;          // this lane's code column
        const float cn = lcn[n];
#pragma unroll
        for (int j = 0; j < 8; ++j) {
            float dl = fmaf(-2.0f, accLo[j], cn);
            float dh = fmaf(-2.0f, accHi[j], cn);
            if (dl < bestLo[j]) { bestLo[j] = dl; bidxLo[j] = n; }
            if (dh < bestHi[j]) { bestHi[j] = dh; bidxHi[j] = n; }
        }
    }

    // --- per-row argmin: reduce across the 16 lanes of each half -----------
#pragma unroll
    for (int s = 1; s < 16; s <<= 1) {
#pragma unroll
        for (int j = 0; j < 8; ++j) {
            float od = __shfl_xor(bestLo[j], s, 32);
            int   oi = __shfl_xor(bidxLo[j], s, 32);
            if (od < bestLo[j] || (od == bestLo[j] && oi < bidxLo[j])) {
                bestLo[j] = od; bidxLo[j] = oi;
            }
            od = __shfl_xor(bestHi[j], s, 32);
            oi = __shfl_xor(bidxHi[j], s, 32);
            if (od < bestHi[j] || (od == bestHi[j] && oi < bidxHi[j])) {
                bestHi[j] = od; bidxHi[j] = oi;
            }
        }
    }
    // lane 0 holds rows {0-7, 16-23}, lane 16 holds rows {8-15, 24-31}.

    // --- gather codebook rows (f32), emit quantized/tokens, exact losses ---
    float lsum = 0.f;
#pragma unroll
    for (int r = 0; r < 32; ++r) {
        const int rr  = r & 15;
        const int src = (rr < 8) ? 0 : 16;
        const int j   = rr & 7;
        const int idx = (r < 16) ? __shfl(bidxLo[j], src, 32)
                                 : __shfl(bidxHi[j], src, 32);
        const int row = rowBase + r;
        const float* q  = cb + (size_t)idx * DDIM + lane * 2;
        const float* xv = x  + (size_t)row * DDIM + lane * 2;
        float q0 = q[0],  q1 = q[1];
        float x0 = xv[0], x1 = xv[1];
        float* o = outq + (size_t)row * DDIM + lane * 2;
        o[0] = q0; o[1] = q1;
        float d0 = q0 - x0, d1 = q1 - x1;
        lsum += d0 * d0 + d1 * d1;
        if (lane == 0) {
            outtok[row] = (float)idx;
            // exact: integer-valued f32 increments < 2^24, order-independent
            atomicAdd(&counts[idx], 1.0f);
        }
    }
    // deterministic in-block loss reduction (no FP atomics)
#pragma unroll
    for (int s = 16; s > 0; s >>= 1) lsum += __shfl_xor(lsum, s, 32);
    if (lane == 0) wsum[wid] = lsum;
    __syncthreads();
    if (tid == 0) {
        float s = 0.f;
#pragma unroll
        for (int w = 0; w < 8; ++w) s += wsum[w];
        lossPart[blockIdx.x] = s;
    }
}

// ---- finalize: losses (fixed-order partial sum) + perplexity ---------------
__global__ void vq_final(const float* __restrict__ counts,
                         const float* __restrict__ lossPart,
                         float* __restrict__ outs) {
    __shared__ float redp[256];
    __shared__ float redl[256];
    const int t = threadIdx.x;
    float sp = 0.f, sl = 0.f;
    for (int k = t; k < KCODES; k += 256) {
        float p = counts[k] * (1.0f / (float)NROWS);
        sp += p * __logf(p + 1e-10f);
    }
    for (int k = t; k < NBLOCK; k += 256) sl += lossPart[k];
    redp[t] = sp;
    redl[t] = sl;
    __syncthreads();
    for (int w = 128; w > 0; w >>= 1) {
        if (t < w) { redp[t] += redp[t + w]; redl[t] += redl[t + w]; }
        __syncthreads();
    }
    if (t == 0) {
        float mse  = redl[0] / (float)(NROWS * DDIM);
        float perp = __expf(-redp[0]);
        // e_latent_loss == q_latent_loss numerically (quantized_st == quantized fwd)
        outs[0] = 1.25f * mse;  // vq_loss = commitment + codebook
        outs[1] = 0.25f * mse;  // commitment_loss
        outs[2] = mse;          // codebook_loss
        outs[3] = perp;         // perplexity
    }
}

extern "C" void kernel_launch(void* const* d_in, const int* in_sizes, int n_in,
                              void* d_out, int out_size, void* d_ws, size_t ws_size,
                              hipStream_t stream) {
    const float* x  = (const float*)d_in[0];   // (64,64,64,64) f32
    const float* cb = (const float*)d_in[1];   // (1024,64) f32
    char* ws = (char*)d_ws;
    unsigned short* cbb = (unsigned short*)ws;
    float* cnorm    = (float*)(ws + WS_CNORM);
    float* counts   = (float*)(ws + WS_COUNTS);
    float* lossPart = (float*)(ws + WS_LOSSP);

    float* out    = (float*)d_out;
    float* outq   = out;                                   // N*D quantized_st
    float* outtok = out + (size_t)NROWS * DDIM;            // N tokens (as f32)
    float* outs   = outtok + NROWS;                        // 4 scalars

    vq_prep_convert<<<256, 256, 0, stream>>>(cb, cbb);
    vq_prep_norm<<<KCODES / 256, 256, 0, stream>>>(cb, cnorm, counts);
    vq_main<<<NBLOCK, 256, 0, stream>>>(x, cb, ws, outq, outtok,
                                        counts, lossPart);
    vq_final<<<1, 256, 0, stream>>>(counts, lossPart, outs);
}
